// lstmModel_79860621902438
// MI455X (gfx1250) — matrix operand order repaired and checked by reading, not compiled, not run on hardware
//
#include <hip/hip_runtime.h>
#include <math.h>

// ---------------- CDNA5 vector types ----------------
typedef __attribute__((ext_vector_type(16))) _Float16 v16h;
typedef __attribute__((ext_vector_type(8)))  _Float16 v8h;
typedef __attribute__((ext_vector_type(8)))  float    v8f;

// ---------------- problem constants ----------------
constexpr int Hh = 150;          // hidden size
constexpr int Tt = 1024;         // time steps
constexpr int Bb = 512;          // batch
constexpr int HP = 160;          // H padded to multiple of 32 (WMMA K)
constexpr int NP = 640;          // 4*H padded (40 N-tiles of 16)
constexpr int BT = 16;           // batch tile per block (one WMMA M tile)
constexpr int BLOCK = 640;       // 20 wave32s -> 2 N-tiles per wave
constexpr int NWAVE = BLOCK / 32;

// ---------------- LDS carve-up (dynamic shared) ----------------
constexpr size_t OFF_W    = 0;                                   // f16 [NP][HP]   204800 B
constexpr size_t OFF_GATE = OFF_W    + (size_t)NP * HP * 2;      // f32 [BT][NP]    40960 B
constexpr size_t OFF_C    = OFF_GATE + (size_t)BT * NP * 4;      // f32 [BT][HP]    10240 B
constexpr size_t OFF_H    = OFF_C    + (size_t)BT * HP * 4;      // f16 [BT][HP]     5120 B
constexpr size_t OFF_WIH  = OFF_H    + (size_t)BT * HP * 2;      // f32 [NP]         2560 B
constexpr size_t OFF_BIAS = OFF_WIH  + (size_t)NP * 4;           // f32 [NP]         2560 B
constexpr size_t OFF_WOUT = OFF_BIAS + (size_t)NP * 4;           // f32 [HP]          640 B
constexpr size_t OFF_SX   = OFF_WOUT + (size_t)HP * 4;           // f32 [BT]           64 B
constexpr size_t SMEM_BYTES = OFF_SX + (size_t)BT * 4;           // = 266944 < 320 KB

__device__ __forceinline__ float sigm(float v) { return 1.f / (1.f + __expf(-v)); }

__global__ __launch_bounds__(BLOCK)
void lstm_wmma_kernel(const float* __restrict__ x,      // [B,T,1]
                      const float* __restrict__ W_ih,   // [4H,1]
                      const float* __restrict__ W_hh,   // [4H,H]
                      const float* __restrict__ b_ih,   // [4H]
                      const float* __restrict__ b_hh,   // [4H]
                      const float* __restrict__ W_out,  // [1,H]
                      const float* __restrict__ b_out,  // [1]
                      float* __restrict__ out)          // [B,T,1]
{
    extern __shared__ char smem[];
    _Float16* Wsh  = (_Float16*)(smem + OFF_W);    // W_hh^T tiles: [n][k], k contiguous
    float*    gate = (float*)   (smem + OFF_GATE); // [BT][NP]
    float*    csh  = (float*)   (smem + OFF_C);    // [BT][HP]
    _Float16* hsh  = (_Float16*)(smem + OFF_H);    // [BT][HP] row-major
    float*    wih  = (float*)   (smem + OFF_WIH);
    float*    bias = (float*)   (smem + OFF_BIAS);
    float*    wout = (float*)   (smem + OFF_WOUT);
    float*    sx   = (float*)   (smem + OFF_SX);

    const int tid = threadIdx.x;
    const int b0  = blockIdx.x * BT;

    // ---------------- prologue: stage weights into LDS as f16, padded ----------------
    {
        // one padded gate-row per thread: n in [0,640)
        const int n    = tid;
        const int gsel = n / HP;        // 0..3 = i,f,g,o
        const int j    = n % HP;        // index inside gate, valid if < 150
        if (j < Hh) {
            const int r = gsel * Hh + j;            // row in global W_hh [600,150]
            const float* src = W_hh + (size_t)r * Hh;
            for (int k = 0; k < Hh; ++k) Wsh[(size_t)n * HP + k] = (_Float16)src[k];
            for (int k = Hh; k < HP; ++k) Wsh[(size_t)n * HP + k] = (_Float16)0.f;
            wih[n]  = W_ih[r];
            bias[n] = b_ih[r] + b_hh[r];
        } else {
            for (int k = 0; k < HP; ++k) Wsh[(size_t)n * HP + k] = (_Float16)0.f;
            wih[n]  = 0.f;
            bias[n] = 0.f;
        }
    }
    if (tid < HP) wout[tid] = (tid < Hh) ? W_out[tid] : 0.f;
    for (int e = tid; e < BT * HP; e += BLOCK) { hsh[e] = (_Float16)0.f; csh[e] = 0.f; }
    __syncthreads();

    const int lane  = tid & 31;
    const int wave  = tid >> 5;
    // Contiguous-K A operand: lanes 0-15 rows M=lane K=0..15, lanes 16-31 same rows K=16..31 (per 32-K chunk)
    const int arow  = lane & 15;
    const int koffA = (lane & 16) ? 16 : 0;
    // Contiguous-K B operand: lanes 0-15 cols N=lane K=0..15, lanes 16-31 K=16..31 (per 32-K chunk)
    const int bcol  = lane & 15;
    const int koffB = (lane & 16) ? 16 : 0;
    // WMMA f32 C/D layout: VGPR j <-> row M=j (+8 for lanes 16-31), col N = lane&15
    const int crow0 = (lane & 16) ? 8 : 0;
    const float bo  = b_out[0];

#pragma unroll 1
    for (int t = 0; t < Tt; ++t) {
        // stage this step's scalar inputs (I == 1)
        if (tid < BT) {
            sx[tid] = x[(size_t)(b0 + tid) * Tt + t];
            if (t + 8 < Tt) __builtin_prefetch(&x[(size_t)(b0 + tid) * Tt + t + 8], 0, 0);
        }

        // -------- recurrent GEMM: gate[BT,NP] = h[BT,HP] @ Wsh^T, f16 WMMA, K=160 --------
#pragma unroll
        for (int nti = 0; nti < 2; ++nti) {
            const int nt = wave * 2 + nti;          // N-tile 0..39
            v8f acc = {};
#pragma unroll
            for (int kc = 0; kc < 5; ++kc) {        // K chunks of 32
                const _Float16* ap = hsh + (size_t)arow * HP + kc * 32 + koffA;
                union { v16h v; v8h h[2]; } A;
                A.h[0] = *(const v8h*)(ap);         // K = base .. base+7
                A.h[1] = *(const v8h*)(ap + 8);     // K = base+8 .. base+15
                const _Float16* bp = Wsh + (size_t)(nt * 16 + bcol) * HP + kc * 32 + koffB;
                const v16h Bv = *(const v16h*)(bp); // 16 contiguous K values of this column
                acc = __builtin_amdgcn_wmma_f32_16x16x32_f16(
                          false, A.v, false, Bv, (short)0, acc, false, false);
            }
#pragma unroll
            for (int jj = 0; jj < 8; ++jj)
                gate[(size_t)(crow0 + jj) * NP + nt * 16 + bcol] = acc[jj];
        }
        __syncthreads();

        // -------- elementwise: activations + c/h update (2560 padded elements) --------
#pragma unroll
        for (int kk = 0; kk < 4; ++kk) {
            const int e   = tid + kk * BLOCK;       // 0..2559
            const int row = e / HP;
            const int j   = e % HP;
            if (j < Hh) {
                const float xs = sx[row];
                const float gi = gate[(size_t)row * NP + 0 * HP + j] + xs * wih[0 * HP + j] + bias[0 * HP + j];
                const float gf = gate[(size_t)row * NP + 1 * HP + j] + xs * wih[1 * HP + j] + bias[1 * HP + j];
                const float gg = gate[(size_t)row * NP + 2 * HP + j] + xs * wih[2 * HP + j] + bias[2 * HP + j];
                const float go = gate[(size_t)row * NP + 3 * HP + j] + xs * wih[3 * HP + j] + bias[3 * HP + j];
                const float iv = sigm(gi);
                const float fv = sigm(gf);
                const float gv = tanhf(gg);
                const float ov = sigm(go);
                const float c  = fv * csh[(size_t)row * HP + j] + iv * gv;
                csh[(size_t)row * HP + j] = c;
                hsh[(size_t)row * HP + j] = (_Float16)(ov * tanhf(c));
            }
        }
        __syncthreads();

        // -------- fused output head: out[b,t] = sigmoid(h . W_out + b_out) --------
        if (wave < BT) {                             // wave == batch row within tile
            float s = 0.f;
#pragma unroll
            for (int j = lane; j < HP; j += 32)      // padding lanes contribute 0
                s += (float)hsh[(size_t)wave * HP + j] * wout[j];
#pragma unroll
            for (int off = 16; off >= 1; off >>= 1)
                s += __shfl_xor(s, off, 32);
            if (lane == 0)
                out[(size_t)(b0 + wave) * Tt + t] = sigm(s + bo);
        }
        // next GEMM only writes `gate`, which every reader released at the barrier above
    }
}

extern "C" void kernel_launch(void* const* d_in, const int* in_sizes, int n_in,
                              void* d_out, int out_size, void* d_ws, size_t ws_size,
                              hipStream_t stream) {
    const float* x     = (const float*)d_in[0];   // [512,1024,1]
    const float* W_ih  = (const float*)d_in[1];   // [600,1]
    const float* W_hh  = (const float*)d_in[2];   // [600,150]
    const float* b_ih  = (const float*)d_in[3];   // [600]
    const float* b_hh  = (const float*)d_in[4];   // [600]
    const float* W_out = (const float*)d_in[5];   // [1,150]
    const float* b_out = (const float*)d_in[6];   // [1]
    float* out = (float*)d_out;                   // [512,1024,1]

    dim3 grid(Bb / BT);     // 32 blocks, each owns 16 batch rows for all 1024 steps
    dim3 block(BLOCK);      // 20 wave32s
    lstm_wmma_kernel<<<grid, block, SMEM_BYTES, stream>>>(
        x, W_ih, W_hh, b_ih, b_hh, W_out, b_out, out);
}
